// Offset_Head_28166395527479
// MI455X (gfx1250) — matrix operand order repaired and checked by reading, not compiled, not run on hardware
//
#include <hip/hip_runtime.h>
#include <stdint.h>
#include <stddef.h>

// ---------------- problem constants ----------------
#define B_     8
#define N_     1024
#define C_     256
#define NHEAD_ 8
#define DH_    32
#define P0_    12288      // 96*128
#define LTOT_  16320      // 12288+3072+768+192
#define DFF_   1024
#define NLAY_  3

#ifndef USE_ASYNC_LDS
#define USE_ASYNC_LDS 1   // stage GEMM B-operand via GLOBAL_LOAD_ASYNC_TO_LDS_B128
#endif

typedef __attribute__((ext_vector_type(16))) __bf16 v16bf;
typedef __attribute__((ext_vector_type(8)))  __bf16 v8bf;
typedef __attribute__((ext_vector_type(8)))  float  v8f;
typedef __attribute__((address_space(3))) uint16_t* lds_u16p;

__device__ __forceinline__ uint16_t f2bf(float f) {
  union { float f; uint32_t u; } x; x.f = f;
  uint32_t r = x.u + 0x7FFFu + ((x.u >> 16) & 1u);   // round-to-nearest-even
  return (uint16_t)(r >> 16);
}

// ---------------- elementwise / setup kernels ----------------

__global__ __launch_bounds__(256) void k_refnorm(const float* __restrict__ tc,
                                                 float* __restrict__ ref, int n) {
  int i = blockIdx.x * 256 + threadIdx.x;
  if (i >= n) return;
  float s = (i & 1) ? (1.0f / 384.0f) : (1.0f / 512.0f);
  ref[i] = fminf(fmaxf(tc[i] * s, 0.0f), 1.0f);
}

__global__ __launch_bounds__(256) void k_initq(const float* __restrict__ src,
                                               float* __restrict__ qf,
                                               uint16_t* __restrict__ qb, int n) {
  int i = blockIdx.x * 256 + threadIdx.x;
  if (i >= n) return;
  float v = src[i];
  qf[i] = v;
  qb[i] = f2bf(v);
}

__global__ __launch_bounds__(256) void k_copy_l0(const float* __restrict__ f,
                                                 float* __restrict__ pyr, int n) {
  int i = blockIdx.x * 256 + threadIdx.x;
  if (i >= n) return;
  int per = P0_ * C_;
  int b = i / per, rem = i - b * per;
  pyr[(size_t)b * LTOT_ * C_ + rem] = f[i];
}

__global__ __launch_bounds__(256) void k_pool(const float* __restrict__ pyr,
                                              float* __restrict__ out,
                                              int srcOff, int dstOff,
                                              int Ws, int Ho, int Wo, int n) {
  int i = blockIdx.x * 256 + threadIdx.x;
  if (i >= n) return;
  int c = i & (C_ - 1);
  int t = i >> 8;
  int x = t % Wo; t /= Wo;
  int y = t % Ho; int b = t / Ho;
  size_t srow = (size_t)b * LTOT_ * C_ + (size_t)(srcOff + (2 * y) * Ws + 2 * x) * C_ + c;
  float v = pyr[srow] + pyr[srow + (size_t)C_] +
            pyr[srow + (size_t)Ws * C_] + pyr[srow + (size_t)(Ws + 1) * C_];
  out[(size_t)b * LTOT_ * C_ + (size_t)(dstOff + y * Wo + x) * C_ + c] = 0.25f * v;
}

__global__ __launch_bounds__(256) void k_cvt_bf16(const float* __restrict__ src,
                                                  uint16_t* __restrict__ dst, int n) {
  int i = blockIdx.x * 256 + threadIdx.x;
  if (i >= n) return;
  dst[i] = f2bf(src[i]);
}

// Pack weight W[K x N] (f32, row-major) into WMMA B-operand tiles.
// tile = nt*(K/32)+kt ; within tile: lane*16 + j ; element j of lane L is
//   W[kt*32 + (L>>4)*16 + j][nt*16 + (L&15)]
__global__ __launch_bounds__(256) void k_pack_b(const float* __restrict__ W,
                                                uint16_t* __restrict__ out,
                                                int K, int Nn) {
  int idx = blockIdx.x * 256 + threadIdx.x;
  if (idx >= K * Nn) return;
  int j    = idx & 15;
  int lane = (idx >> 4) & 31;
  int tile = idx >> 9;
  int ktiles = K >> 5;
  int kt = tile % ktiles;
  int nt = tile / ktiles;
  int k = kt * 32 + (lane >> 4) * 16 + j;
  int n = nt * 16 + (lane & 15);
  out[idx] = f2bf(W[(size_t)k * Nn + n]);
}

// ---------------- bf16 WMMA GEMM with async-LDS B staging ----------------
// Out[M x Nn] = A[M x K](bf16) @ B(packed bf16) + bias.
// grid = (M/256, Nn/64), block = 256 (8 waves). Wave tile: 32(M) x 64(N)
// = 8 accumulators, 8 WMMAs per K-step of 32.
// B is staged chunk-wise (8 K-tiles = 32KB) into LDS via async b128 copies.
#define KCH 8
template <int RELU, int SF32, int SB16>
__global__ __launch_bounds__(256) void k_gemm_wmma(const uint16_t* __restrict__ A,
                                                   const uint16_t* __restrict__ Bp,
                                                   const float* __restrict__ bias,
                                                   float* __restrict__ Cf,
                                                   uint16_t* __restrict__ Cb,
                                                   int M, int K, int Nn) {
  __shared__ uint16_t smem[KCH * 4 * 512] __attribute__((aligned(128)));  // 32KB

  const int tid  = threadIdx.x;
  const int wave = tid >> 5;
  const int lane = tid & 31;
  const int hs   = lane >> 4;          // K-half select for A/B lane layout
  const int nl   = lane & 15;
  const int m0   = blockIdx.x * 256 + wave * 32;
  const int n0   = blockIdx.y * 64;
  const int ktiles = K >> 5;

  const uint16_t* __restrict__ arow0 = A + (size_t)(m0 + nl) * K;
  const uint16_t* __restrict__ arow1 = A + (size_t)(m0 + 16 + nl) * K;

  v8f cacc[8];
  #pragma unroll
  for (int t = 0; t < 8; ++t) cacc[t] = (v8f){};

  const uint64_t bp64 = (uint64_t)(uintptr_t)Bp;
  lds_u16p lbase = (lds_u16p)smem;

  for (int kc = 0; kc < ktiles; kc += KCH) {
    // ---- cooperative staging of B chunk: 4 runs of 8KB (one per N-tile) ----
    #pragma unroll
    for (int nt = 0; nt < 4; ++nt) {
      const uint32_t gb = (uint32_t)((((size_t)((n0 >> 4) + nt)) * ktiles + kc) * 1024)
                        + (uint32_t)tid * 32u;                    // global byte offset
      lds_u16p ld = lbase + (nt * KCH * 512) + tid * 16;          // halfs
#if USE_ASYNC_LDS
      asm volatile("global_load_async_to_lds_b128 %0, %1, %2\n\t"
                   "global_load_async_to_lds_b128 %0, %1, %2 offset:16"
                   :: "v"(ld), "v"(gb), "s"(bp64) : "memory");
#else
      const v8bf* gsrc = (const v8bf*)(Bp + (size_t)gb / 2);
      *(__attribute__((address_space(3))) v8bf*)(ld)     = gsrc[0];
      *(__attribute__((address_space(3))) v8bf*)(ld + 8) = gsrc[1];
#endif
    }
#if USE_ASYNC_LDS
    asm volatile("s_wait_asynccnt 0x0" ::: "memory");
#endif
    __syncthreads();

    // ---- compute on the staged chunk ----
    #pragma unroll
    for (int ktl = 0; ktl < KCH; ++ktl) {
      const int kbase = (kc + ktl) * 32 + hs * 16;
      v8bf a0lo = *(const v8bf*)(arow0 + kbase);
      v8bf a0hi = *(const v8bf*)(arow0 + kbase + 8);
      v8bf a1lo = *(const v8bf*)(arow1 + kbase);
      v8bf a1hi = *(const v8bf*)(arow1 + kbase + 8);
      v16bf a0, a1;
      #pragma unroll
      for (int i = 0; i < 8; ++i) {
        a0[i] = a0lo[i]; a0[8 + i] = a0hi[i];
        a1[i] = a1lo[i]; a1[8 + i] = a1hi[i];
      }
      v16bf bt[4];
      #pragma unroll
      for (int nt = 0; nt < 4; ++nt)
        bt[nt] = *(const v16bf*)(smem + (nt * KCH + ktl) * 512 + lane * 16);

      #pragma unroll
      for (int nt = 0; nt < 4; ++nt) {
        cacc[nt]     = __builtin_amdgcn_wmma_f32_16x16x32_bf16(false, a0, false, bt[nt], (short)0, cacc[nt],     false, false);
        cacc[4 + nt] = __builtin_amdgcn_wmma_f32_16x16x32_bf16(false, a1, false, bt[nt], (short)0, cacc[4 + nt], false, false);
      }
    }
    __syncthreads();   // safe to overwrite smem with next chunk
  }

  // ---- epilogue ----
  #pragma unroll
  for (int nt = 0; nt < 4; ++nt) {
    const int col = n0 + nt * 16 + nl;
    const float bv = bias[col];
    #pragma unroll
    for (int mt = 0; mt < 2; ++mt) {
      #pragma unroll
      for (int r = 0; r < 8; ++r) {
        const int row = m0 + mt * 16 + hs * 8 + r;
        float val = cacc[mt * 4 + nt][r] + bv;
        if (RELU) val = fmaxf(val, 0.0f);
        if (SF32) Cf[(size_t)row * Nn + col] = val;
        if (SB16) Cb[(size_t)row * Nn + col] = f2bf(val);
      }
    }
  }
}

// ---------------- deformable sampling (softmax fused) ----------------
// one wave per (b, n, head); lane = channel d (coalesced 128B taps)
__global__ __launch_bounds__(256) void k_sample(const float* __restrict__ v,     // [B,16320,8,32]
                                                const float* __restrict__ ref,   // [B,N,2]
                                                const float* __restrict__ off,   // [B,N,256]
                                                const float* __restrict__ awl,   // [B,N,128]
                                                uint16_t* __restrict__ outb) {   // [B,N,256] bf16
  const int wid  = (blockIdx.x * 256 + threadIdx.x) >> 5;
  const int lane = threadIdx.x & 31;
  const int h = wid & (NHEAD_ - 1);
  const int n = (wid >> 3) & (N_ - 1);
  const int b = wid >> 13;
  const size_t row = (size_t)b * N_ + n;

  const float rx = ref[row * 2 + 0];
  const float ry = ref[row * 2 + 1];

  const float* awp = awl + row * 128 + h * 16;
  float e = (lane < 16) ? awp[lane] : -1e30f;
  float mx = e;
  #pragma unroll
  for (int o = 16; o > 0; o >>= 1) mx = fmaxf(mx, __shfl_xor(mx, o));
  float ex = (lane < 16) ? __expf(e - mx) : 0.0f;
  float sm = ex;
  #pragma unroll
  for (int o = 16; o > 0; o >>= 1) sm += __shfl_xor(sm, o);
  const float wn = ex / sm;

  const float* offp = off + row * 256 + h * 32;

  const int lvlH[4] = {96, 48, 24, 12};
  const int lvlW[4] = {128, 64, 32, 16};
  const int lvlO[4] = {0, 12288, 15360, 16128};

  float acc = 0.0f;
  #pragma unroll
  for (int l = 0; l < 4; ++l) {
    const int H = lvlH[l], W = lvlW[l];
    const int base = b * LTOT_ + lvlO[l];
    const float fW = (float)W, fH = (float)H;
    #pragma unroll
    for (int p = 0; p < 4; ++p) {
      const float ox = offp[l * 8 + p * 2 + 0];
      const float oy = offp[l * 8 + p * 2 + 1];
      const float x = (rx + ox / fW) * fW - 0.5f;
      const float y = (ry + oy / fH) * fH - 0.5f;
      const float x0f = floorf(x), y0f = floorf(y);
      const float tx = x - x0f, ty = y - y0f;
      const int x0 = (int)x0f, y0 = (int)y0f;
      const float w = __shfl(wn, l * 4 + p);

      float s00 = 0.f, s01 = 0.f, s10 = 0.f, s11 = 0.f;
      {
        const int x1 = x0 + 1, y1 = y0 + 1;
        const bool vx0 = (x0 >= 0) & (x0 < W), vx1 = (x1 >= 0) & (x1 < W);
        const bool vy0 = (y0 >= 0) & (y0 < H), vy1 = (y1 >= 0) & (y1 < H);
        if (vy0) {
          const size_t rb = (((size_t)(base + y0 * W)) * NHEAD_ + h) * DH_ + lane;
          if (vx0) s00 = v[rb + (size_t)x0 * (NHEAD_ * DH_)];
          if (vx1) s01 = v[rb + (size_t)x1 * (NHEAD_ * DH_)];
        }
        if (vy1) {
          const size_t rb = (((size_t)(base + y1 * W)) * NHEAD_ + h) * DH_ + lane;
          if (vx0) s10 = v[rb + (size_t)x0 * (NHEAD_ * DH_)];
          if (vx1) s11 = v[rb + (size_t)x1 * (NHEAD_ * DH_)];
        }
      }
      const float bi = s00 * (1.f - tx) * (1.f - ty) + s01 * tx * (1.f - ty)
                     + s10 * (1.f - tx) * ty + s11 * tx * ty;
      acc += w * bi;
    }
  }
  outb[(row * NHEAD_ + h) * DH_ + lane] = f2bf(acc);
}

// ---------------- fused residual + layernorm (writes f32 + bf16) ----------------
__global__ __launch_bounds__(256) void k_res_ln(const float* __restrict__ x,
                                                const float* __restrict__ r,
                                                const float* __restrict__ g,
                                                const float* __restrict__ bb,
                                                float* __restrict__ y,
                                                uint16_t* __restrict__ yb) {
  const int wave = threadIdx.x >> 5;
  const int lane = threadIdx.x & 31;
  const size_t row = (size_t)blockIdx.x * 8 + wave;
  const float* xr = x + row * C_;
  const float* rr = r + row * C_;
  float v[8], s = 0.f, s2 = 0.f;
  #pragma unroll
  for (int j = 0; j < 8; ++j) {
    const int c = lane + 32 * j;
    const float t = xr[c] + rr[c];
    v[j] = t; s += t; s2 += t * t;
  }
  #pragma unroll
  for (int o = 16; o > 0; o >>= 1) { s += __shfl_xor(s, o); s2 += __shfl_xor(s2, o); }
  const float mean = s * (1.0f / C_);
  const float var  = s2 * (1.0f / C_) - mean * mean;
  const float inv  = rsqrtf(var + 1e-5f);
  #pragma unroll
  for (int j = 0; j < 8; ++j) {
    const int c = lane + 32 * j;
    const float o = (v[j] - mean) * inv * g[c] + bb[c];
    y[row * C_ + c] = o;
    yb[row * C_ + c] = f2bf(o);
  }
}

// ---------------- offset head: LN -> linear(256->2) -> tanh*4 ----------------
__global__ __launch_bounds__(256) void k_offhead(const float* __restrict__ q,
                                                 const float* __restrict__ g,
                                                 const float* __restrict__ bb,
                                                 const float* __restrict__ W,   // [256,2]
                                                 const float* __restrict__ bo,  // [2]
                                                 float* __restrict__ out, int layer) {
  const int wave = threadIdx.x >> 5;
  const int lane = threadIdx.x & 31;
  const size_t row = (size_t)blockIdx.x * 8 + wave;   // b*N + n
  const float* qr = q + row * C_;
  float v[8], s = 0.f, s2 = 0.f;
  #pragma unroll
  for (int j = 0; j < 8; ++j) {
    const float t = qr[lane + 32 * j];
    v[j] = t; s += t; s2 += t * t;
  }
  #pragma unroll
  for (int o = 16; o > 0; o >>= 1) { s += __shfl_xor(s, o); s2 += __shfl_xor(s2, o); }
  const float mean = s * (1.0f / C_);
  const float inv  = rsqrtf(s2 * (1.0f / C_) - mean * mean + 1e-5f);
  float d0 = 0.f, d1 = 0.f;
  #pragma unroll
  for (int j = 0; j < 8; ++j) {
    const int c = lane + 32 * j;
    const float t = (v[j] - mean) * inv * g[c] + bb[c];
    d0 += t * W[c * 2 + 0];
    d1 += t * W[c * 2 + 1];
  }
  #pragma unroll
  for (int o = 16; o > 0; o >>= 1) { d0 += __shfl_xor(d0, o); d1 += __shfl_xor(d1, o); }
  if (lane == 0) {
    const int b = (int)(row >> 10), n = (int)(row & (N_ - 1));
    const size_t o = (((size_t)b * NLAY_ + layer) * N_ + n) * 2;
    out[o + 0] = tanhf(d0 + bo[0]) * 4.0f;
    out[o + 1] = tanhf(d1 + bo[1]) * 4.0f;
  }
}

// ---------------- host-side orchestration ----------------
extern "C" void kernel_launch(void* const* d_in, const int* in_sizes, int n_in,
                              void* d_out, int out_size, void* d_ws, size_t ws_size,
                              hipStream_t stream) {
  (void)in_sizes; (void)n_in; (void)out_size; (void)ws_size;
  const float* q_t = (const float*)d_in[0];
  const float* f_t = (const float*)d_in[1];
  const float* tc  = (const float*)d_in[2];
  // params: layers[l] has 16 leaves starting at 3 + 16*l:
  //  +0 w_value +1 b_value +2 w_so +3 b_so +4 w_aw +5 b_aw +6 w_op +7 b_op
  //  +8 ln1_g +9 ln1_b +10 w_f1 +11 b_f1 +12 w_f2 +13 b_f2 +14 ln2_g +15 ln2_b
  const int PB = 3;
  const int off_ln_g = PB + 16 * NLAY_ + 0;
  const int off_ln_b = PB + 16 * NLAY_ + 1;
  const int off_w    = PB + 16 * NLAY_ + 2;
  const int off_b    = PB + 16 * NLAY_ + 3;

  char* wp = (char*)d_ws;
  auto alloc = [&](size_t bytes) -> void* {
    void* r = (void*)wp;
    wp += (bytes + 255) & ~(size_t)255;
    return r;
  };

  const size_t MV = (size_t)B_ * LTOT_;        // 130560
  const size_t MQ = (size_t)B_ * N_;           // 8192

  float*    vbuf   = (float*)alloc(MV * C_ * 4);     // pyramid f32, then per-layer value v
  uint16_t* pyr_b  = (uint16_t*)alloc(MV * C_ * 2);  // pyramid bf16 (GEMM A)
  float*    qf     = (float*)alloc(MQ * C_ * 4);
  uint16_t* qb     = (uint16_t*)alloc(MQ * C_ * 2);
  float*    offb   = (float*)alloc(MQ * C_ * 4);     // sampling offsets
  float*    awb    = (float*)alloc(MQ * 128 * 4);    // attn logits
  uint16_t* attnb  = (uint16_t*)alloc(MQ * C_ * 2);  // sampled attn (bf16 GEMM A)
  float*    proj   = (float*)alloc(MQ * C_ * 4);     // out_proj / ffn2 result
  uint16_t* hb     = (uint16_t*)alloc(MQ * DFF_ * 2);// ffn hidden bf16
  float*    refn   = (float*)alloc(MQ * 2 * 4);

  static const int wIdx[6] = {0, 2, 4, 6, 10, 12};
  static const int wK[6]   = {C_, C_, C_, C_, C_, DFF_};
  static const int wN[6]   = {C_, C_, 128, C_, DFF_, C_};
  uint16_t* packed[NLAY_][6];
  for (int l = 0; l < NLAY_; ++l)
    for (int i = 0; i < 6; ++i)
      packed[l][i] = (uint16_t*)alloc((size_t)wK[i] * wN[i] * 2);

  // ---- setup ----
  k_refnorm<<<(int)((MQ * 2 + 255) / 256), 256, 0, stream>>>(tc, refn, (int)(MQ * 2));
  k_initq<<<(int)((MQ * C_ + 255) / 256), 256, 0, stream>>>(q_t, qf, qb, (int)(MQ * C_));
  k_copy_l0<<<(B_ * P0_ * C_ + 255) / 256, 256, 0, stream>>>(f_t, vbuf, B_ * P0_ * C_);
  k_pool<<<(B_ * 48 * 64 * C_) / 256, 256, 0, stream>>>(vbuf, vbuf, 0,     12288, 128, 48, 64, B_ * 48 * 64 * C_);
  k_pool<<<(B_ * 24 * 32 * C_) / 256, 256, 0, stream>>>(vbuf, vbuf, 12288, 15360, 64,  24, 32, B_ * 24 * 32 * C_);
  k_pool<<<(B_ * 12 * 16 * C_) / 256, 256, 0, stream>>>(vbuf, vbuf, 15360, 16128, 32,  12, 16, B_ * 12 * 16 * C_);
  k_cvt_bf16<<<(int)((MV * C_ + 255) / 256), 256, 0, stream>>>(vbuf, pyr_b, (int)(MV * C_));

  for (int l = 0; l < NLAY_; ++l)
    for (int i = 0; i < 6; ++i)
      k_pack_b<<<(wK[i] * wN[i]) / 256, 256, 0, stream>>>(
          (const float*)d_in[PB + 16 * l + wIdx[i]], packed[l][i], wK[i], wN[i]);

  // ---- layers ----
  for (int l = 0; l < NLAY_; ++l) {
    const int base = PB + 16 * l;
    const float* b_v  = (const float*)d_in[base + 1];
    const float* b_so = (const float*)d_in[base + 3];
    const float* b_aw = (const float*)d_in[base + 5];
    const float* b_op = (const float*)d_in[base + 7];
    const float* ln1g = (const float*)d_in[base + 8];
    const float* ln1b = (const float*)d_in[base + 9];
    const float* b_f1 = (const float*)d_in[base + 11];
    const float* b_f2 = (const float*)d_in[base + 13];
    const float* ln2g = (const float*)d_in[base + 14];
    const float* ln2b = (const float*)d_in[base + 15];

    // v = value_proj(pyramid)   M=130560 K=256 N=256
    k_gemm_wmma<0, 1, 0><<<dim3((int)(MV / 256), C_ / 64), 256, 0, stream>>>(
        pyr_b, packed[l][0], b_v, vbuf, nullptr, (int)MV, C_, C_);
    // sampling offsets          M=8192 K=256 N=256
    k_gemm_wmma<0, 1, 0><<<dim3((int)(MQ / 256), C_ / 64), 256, 0, stream>>>(
        qb, packed[l][1], b_so, offb, nullptr, (int)MQ, C_, C_);
    // attn logits               M=8192 K=256 N=128
    k_gemm_wmma<0, 1, 0><<<dim3((int)(MQ / 256), 128 / 64), 256, 0, stream>>>(
        qb, packed[l][2], b_aw, awb, nullptr, (int)MQ, C_, 128);
    // deformable sampling + softmax -> attn (bf16)
    k_sample<<<(B_ * N_ * NHEAD_) / 8, 256, 0, stream>>>(vbuf, refn, offb, awb, attnb);
    // out_proj                  M=8192 K=256 N=256
    k_gemm_wmma<0, 1, 0><<<dim3((int)(MQ / 256), C_ / 64), 256, 0, stream>>>(
        attnb, packed[l][3], b_op, proj, nullptr, (int)MQ, C_, C_);
    // q = LN(q + attn)
    k_res_ln<<<(int)(MQ / 8), 256, 0, stream>>>(qf, proj, ln1g, ln1b, qf, qb);
    // ffn1 + relu -> bf16       M=8192 K=256 N=1024
    k_gemm_wmma<1, 0, 1><<<dim3((int)(MQ / 256), DFF_ / 64), 256, 0, stream>>>(
        qb, packed[l][4], b_f1, nullptr, hb, (int)MQ, C_, DFF_);
    // ffn2                      M=8192 K=1024 N=256
    k_gemm_wmma<0, 1, 0><<<dim3((int)(MQ / 256), C_ / 64), 256, 0, stream>>>(
        hb, packed[l][5], b_f2, proj, nullptr, (int)MQ, DFF_, C_);
    // q = LN(q + ffn)
    k_res_ln<<<(int)(MQ / 8), 256, 0, stream>>>(qf, proj, ln2g, ln2b, qf, qb);
    // offset head -> output slice
    k_offhead<<<(int)(MQ / 8), 256, 0, stream>>>(
        qf, (const float*)d_in[off_ln_g], (const float*)d_in[off_ln_b],
        (const float*)d_in[off_w], (const float*)d_in[off_b], (float*)d_out, l);
  }
}
